// MultiHeadAttention_64020782514538
// MI455X (gfx1250) — matrix operand order repaired and checked by reading, not compiled, not run on hardware
//
#include <hip/hip_runtime.h>
#include <hip/hip_bf16.h>

// ---------------------------------------------------------------------------
// MHA forward for MI455X (gfx1250): bf16 WMMA everywhere, f32 accumulation.
//   1) convert x, Wq/Wk/Wv/Wo f32 -> bf16 (workspace)
//   2) Q/K/V projections: bf16 GEMM, double-buffered async global->LDS staging
//   3) flash-attention per (b,h) with online softmax, WMMA for QK^T and PV
//   4) output projection -> f32 d_out
// ---------------------------------------------------------------------------

typedef __attribute__((ext_vector_type(16))) __bf16 v16bf;
typedef __attribute__((ext_vector_type(8)))  __bf16 v8bf;
typedef __attribute__((ext_vector_type(8)))  float  v8f;

union FragA {           // 16x32 bf16 A fragment: two 8-element halves per lane
    v16bf v;
    v8bf  h[2];
};

__device__ __forceinline__ unsigned short f2b(float f) {
    unsigned int u = __float_as_uint(f);
    unsigned int r = u + 0x7FFFu + ((u >> 16) & 1u);   // round-to-nearest-even
    return (unsigned short)(r >> 16);
}

// ---- async global -> LDS copy (16 bytes), guarded so compile never fails ---
#if defined(__AMDGCN__) && __has_builtin(__builtin_amdgcn_global_load_async_to_lds_b128)
#define ASYNC_LDS_CP 1
typedef int async_v4i __attribute__((vector_size(16)));
typedef __attribute__((address_space(1))) async_v4i* gbl_v4i_p;
typedef __attribute__((address_space(3))) async_v4i* lds_v4i_p;
__device__ __forceinline__ void async_cp16(const unsigned short* g, unsigned short* l) {
    __builtin_amdgcn_global_load_async_to_lds_b128(
        (gbl_v4i_p)(g), (lds_v4i_p)(l), 0, 0);
}
__device__ __forceinline__ void async_wait0() {
    asm volatile("s_wait_asynccnt 0x0" ::: "memory");
}
#else
#define ASYNC_LDS_CP 0
__device__ __forceinline__ void async_cp16(const unsigned short* g, unsigned short* l) {
    *(uint4*)l = *(const uint4*)g;
}
__device__ __forceinline__ void async_wait0() {}
#endif

// ---------------------------------------------------------------------------
__global__ void f32_to_bf16_kernel(const float* __restrict__ src,
                                   unsigned short* __restrict__ dst, size_t n) {
    size_t i = (size_t)blockIdx.x * blockDim.x + threadIdx.x;
    size_t stride = (size_t)gridDim.x * blockDim.x;
    for (; i < n; i += stride) dst[i] = f2b(src[i]);
}

// ---------------------------------------------------------------------------
// y[m,n] = sum_k A[m,k] * W[n,k]   (A: [M,K] bf16, W: [N,K] bf16 row-major)
// Block: 256 thr = 8 waves (4 M x 2 N), block tile 128x64, wave tile 32x32.
// K-step 32, double-buffered async staging: A tile 128x32, B tile 64x32.
template <bool OUT_F32>
__global__ __launch_bounds__(256)
void gemm_bf16_kernel(const unsigned short* __restrict__ A,
                      const unsigned short* __restrict__ W,
                      void* __restrict__ outp, int M, int N, int K) {
    __shared__ __align__(32) unsigned short LA[2][128 * 32];
    __shared__ __align__(32) unsigned short LB[2][64 * 32];

    const int tid  = threadIdx.x;
    const int lane = tid & 31;
    const int wave = tid >> 5;
    const int wm = wave >> 1, wn = wave & 1;
    const int m0b = blockIdx.y * 128;
    const int n0b = blockIdx.x * 64;
    const int lo = lane & 15;   // row (A) / col (B/C) within 16-tile
    const int hi = lane >> 4;   // K-half selector: k = hi*16 .. hi*16+15 (contiguous)

    // --- stage one K-tile (A: 512 16B segs, B: 256 16B segs) into LDS ---
    auto stage = [&](int kk, int buf) {
#pragma unroll
        for (int r = 0; r < 2; ++r) {           // A tile: 2 segs per thread
            int id = tid + r * 256;
            int row = id >> 2, seg = id & 3;
            async_cp16(A + (size_t)(m0b + row) * K + kk + seg * 8,
                       &LA[buf][row * 32 + seg * 8]);
        }
        {                                        // B tile: 1 seg per thread
            int row = tid >> 2, seg = tid & 3;
            async_cp16(W + (size_t)(n0b + row) * K + kk + seg * 8,
                       &LB[buf][row * 32 + seg * 8]);
        }
    };

    v8f acc[2][2] = {};
    const int nsteps = K / 32;

    stage(0, 0);
    for (int step = 0; step < nsteps; ++step) {
        const int buf = step & 1;
        async_wait0();           // this thread's stage(step) landed in LDS
        __syncthreads();         // everyone's stage done; compute(step-1) done
        if (step + 1 < nsteps) stage((step + 1) * 32, buf ^ 1);

        const unsigned short* la = LA[buf];
        const unsigned short* lb = LB[buf];
        FragA a[2];
#pragma unroll
        for (int i = 0; i < 2; ++i) {
            const unsigned short* ap = la + (wm * 32 + i * 16 + lo) * 32;
            a[i].h[0] = *(const v8bf*)(ap + hi * 16);      // K = hi*16 .. +7
            a[i].h[1] = *(const v8bf*)(ap + hi * 16 + 8);  // K = hi*16+8 ..
        }
        v16bf bfr[2];
#pragma unroll
        for (int j = 0; j < 2; ++j)
            bfr[j] = *(const v16bf*)(lb + (wn * 32 + j * 16 + lo) * 32 + hi * 16);
#pragma unroll
        for (int i = 0; i < 2; ++i)
#pragma unroll
            for (int j = 0; j < 2; ++j)
                acc[i][j] = __builtin_amdgcn_wmma_f32_16x16x32_bf16(
                    false, a[i].v, false, bfr[j], (short)0, acc[i][j],
                    false, false);
    }

#pragma unroll
    for (int i = 0; i < 2; ++i)
#pragma unroll
        for (int j = 0; j < 2; ++j)
#pragma unroll
            for (int r = 0; r < 8; ++r) {
                int row = m0b + wm * 32 + i * 16 + r + 8 * hi;  // C layout
                int col = n0b + wn * 32 + j * 16 + lo;
                float v = acc[i][j][r];
                if (OUT_F32)
                    ((float*)outp)[(size_t)row * N + col] = v;
                else
                    ((unsigned short*)outp)[(size_t)row * N + col] = f2b(v);
            }
}

// ---------------------------------------------------------------------------
// Flash attention: one block = 64 query rows of one (b,h); 4 waves x 16 rows.
// Q/K/V are [B, T, H, Dh] bf16.  Output Ob same layout ( == [B,T,C] flat).
__global__ __launch_bounds__(128)
void attn_kernel(const unsigned short* __restrict__ Qb,
                 const unsigned short* __restrict__ Kb,
                 const unsigned short* __restrict__ Vb,
                 const int* __restrict__ mask,
                 unsigned short* __restrict__ Ob) {
    constexpr int T = 2048, H = 16, Dh = 64;

    __shared__ __align__(32) unsigned short VT[64 * 64];      // V chunk, [d][kv]
    __shared__ __align__(32) unsigned short Ps[4][16 * 64];   // P per wave

    const int lane = threadIdx.x & 31;
    const int wave = threadIdx.x >> 5;
    const int lo = lane & 15, hi = lane >> 4;
    const int b = blockIdx.y / H, h = blockIdx.y % H;
    const int q0 = blockIdx.x * 64 + wave * 16;

    // Persistent Q fragments (dbase 0 and 32)
    FragA qa[2];
    {
        const unsigned short* qp = Qb + (((size_t)b * T + (q0 + lo)) * H + h) * Dh;
#pragma unroll
        for (int d = 0; d < 2; ++d) {
            qa[d].h[0] = *(const v8bf*)(qp + d * 32 + hi * 16);
            qa[d].h[1] = *(const v8bf*)(qp + d * 32 + hi * 16 + 8);
        }
    }

    float mrow[8], lrow[8];
    v8f oacc[4] = {};
#pragma unroll
    for (int r = 0; r < 8; ++r) { mrow[r] = -3.0e38f; lrow[r] = 0.0f; }

    const int svkv = threadIdx.x >> 1;          // V staging: row per thread pair
    const int svd  = (threadIdx.x & 1) * 32;    // each covers 32 of 64 d values

    for (int kv0 = 0; kv0 < T; kv0 += 64) {
        // ---- prefetch next chunk's K/V rows into cache ----
        if (kv0 + 64 < T) {
            const size_t nrow = ((size_t)b * T + kv0 + 64 + svkv) * H + h;
            __builtin_prefetch(Vb + nrow * Dh, 0, 0);
            __builtin_prefetch(Kb + nrow * Dh, 0, 0);
        }

        // ---- stage V chunk transposed into LDS: VT[d][kv] ----
        __syncthreads();
        {
            const unsigned short* vp =
                Vb + (((size_t)b * T + kv0 + svkv) * H + h) * Dh + svd;
            union { uint4 u; unsigned short s[8]; } seg;
#pragma unroll
            for (int j = 0; j < 4; ++j) {
                seg.u = *(const uint4*)(vp + j * 8);
#pragma unroll
                for (int e = 0; e < 8; ++e)
                    VT[(svd + j * 8 + e) * 64 + svkv] = seg.s[e];
            }
        }
        __syncthreads();

        // ---- S = Q K^T (16 x 64), f32 ----
        v8f s[4];
#pragma unroll
        for (int nt = 0; nt < 4; ++nt) {
            v8f accs = {};
#pragma unroll
            for (int ds = 0; ds < 2; ++ds) {
                const unsigned short* kp =
                    Kb + (((size_t)b * T + kv0 + nt * 16 + lo) * H + h) * Dh +
                    ds * 32 + hi * 16;
                v16bf kfrag = *(const v16bf*)kp;
                accs = __builtin_amdgcn_wmma_f32_16x16x32_bf16(
                    false, qa[ds].v, false, kfrag, (short)0, accs, false, false);
            }
            s[nt] = accs;
        }

        // ---- scale + mask ----
#pragma unroll
        for (int nt = 0; nt < 4; ++nt)
#pragma unroll
            for (int r = 0; r < 8; ++r) {
                int q = q0 + r + 8 * hi;
                int kc = kv0 + nt * 16 + lo;
                float val = s[nt][r] * 0.125f;                 // 1/sqrt(64)
                if (mask[((size_t)b * T + q) * T + kc] == 0) val = -1.0e9f;
                s[nt][r] = val;
            }

        // ---- online softmax update (reductions within 16-lane halves) ----
#pragma unroll
        for (int r = 0; r < 8; ++r) {
            float rmax = fmaxf(fmaxf(s[0][r], s[1][r]), fmaxf(s[2][r], s[3][r]));
            rmax = fmaxf(rmax, __shfl_xor(rmax, 1, 32));
            rmax = fmaxf(rmax, __shfl_xor(rmax, 2, 32));
            rmax = fmaxf(rmax, __shfl_xor(rmax, 4, 32));
            rmax = fmaxf(rmax, __shfl_xor(rmax, 8, 32));
            float newm = fmaxf(mrow[r], rmax);
            float corr = __expf(mrow[r] - newm);
            float rsum = 0.0f;
#pragma unroll
            for (int nt = 0; nt < 4; ++nt) {
                float p = __expf(s[nt][r] - newm);
                s[nt][r] = p;
                rsum += p;
            }
            rsum += __shfl_xor(rsum, 1, 32);
            rsum += __shfl_xor(rsum, 2, 32);
            rsum += __shfl_xor(rsum, 4, 32);
            rsum += __shfl_xor(rsum, 8, 32);
            lrow[r] = lrow[r] * corr + rsum;
            mrow[r] = newm;
#pragma unroll
            for (int dt = 0; dt < 4; ++dt) oacc[dt][r] *= corr;
        }

        // ---- P (C-layout) -> LDS bf16 row-major [16 x 64] ----
        unsigned short* pw = Ps[wave];
#pragma unroll
        for (int nt = 0; nt < 4; ++nt)
#pragma unroll
            for (int r = 0; r < 8; ++r)
                pw[(r + 8 * hi) * 64 + nt * 16 + lo] = f2b(s[nt][r]);

        // ---- O += P @ Vchunk  (K dim = 64 -> two K=32 WMMA steps) ----
#pragma unroll
        for (int ks = 0; ks < 2; ++ks) {
            FragA pa;
            pa.h[0] = *(const v8bf*)(pw + lo * 64 + ks * 32 + hi * 16);
            pa.h[1] = *(const v8bf*)(pw + lo * 64 + ks * 32 + hi * 16 + 8);
#pragma unroll
            for (int dt = 0; dt < 4; ++dt) {
                const unsigned short* vp =
                    VT + (dt * 16 + lo) * 64 + ks * 32 + hi * 16;
                v16bf vfrag = *(const v16bf*)vp;
                oacc[dt] = __builtin_amdgcn_wmma_f32_16x16x32_bf16(
                    false, pa.v, false, vfrag, (short)0, oacc[dt], false, false);
            }
        }
    }

    // ---- finalize: O / l, write bf16 [B,T,H,Dh] ----
#pragma unroll
    for (int dt = 0; dt < 4; ++dt)
#pragma unroll
        for (int r = 0; r < 8; ++r) {
            int q = q0 + r + 8 * hi;
            int d = dt * 16 + lo;
            float v = oacc[dt][r] / lrow[r];
            Ob[(((size_t)b * T + q) * H + h) * Dh + d] = f2b(v);
        }
}

// ---------------------------------------------------------------------------
extern "C" void kernel_launch(void* const* d_in, const int* in_sizes, int n_in,
                              void* d_out, int out_size, void* d_ws, size_t ws_size,
                              hipStream_t stream) {
    (void)in_sizes; (void)n_in; (void)out_size; (void)ws_size;

    const float* x  = (const float*)d_in[0];
    const int* mask = (const int*)d_in[1];
    const float* Wq = (const float*)d_in[2];
    const float* Wk = (const float*)d_in[3];
    const float* Wv = (const float*)d_in[4];
    const float* Wo = (const float*)d_in[5];

    const int B = 4, T = 2048, C = 1024, H = 16;
    const size_t nX = (size_t)B * T * C;   // 8388608
    const size_t nW = (size_t)C * C;       // 1048576

    unsigned short* ws  = (unsigned short*)d_ws;
    unsigned short* xb  = ws;              // x bf16
    unsigned short* wqb = xb  + nX;
    unsigned short* wkb = wqb + nW;
    unsigned short* wvb = wkb + nW;
    unsigned short* wob = wvb + nW;
    unsigned short* qb  = wob + nW;        // Q [B,T,H,Dh]
    unsigned short* kb  = qb  + nX;        // K
    unsigned short* vb  = kb  + nX;        // V
    unsigned short* ob  = vb  + nX;        // attn out bf16 [B,T,C]

    f32_to_bf16_kernel<<<2048, 256, 0, stream>>>(x,  xb,  nX);
    f32_to_bf16_kernel<<<512,  256, 0, stream>>>(Wq, wqb, nW);
    f32_to_bf16_kernel<<<512,  256, 0, stream>>>(Wk, wkb, nW);
    f32_to_bf16_kernel<<<512,  256, 0, stream>>>(Wv, wvb, nW);
    f32_to_bf16_kernel<<<512,  256, 0, stream>>>(Wo, wob, nW);

    dim3 gg(C / 64, (B * T) / 128);        // (16, 64)
    gemm_bf16_kernel<false><<<gg, 256, 0, stream>>>(xb, wqb, qb, B * T, C, C);
    gemm_bf16_kernel<false><<<gg, 256, 0, stream>>>(xb, wkb, kb, B * T, C, C);
    gemm_bf16_kernel<false><<<gg, 256, 0, stream>>>(xb, wvb, vb, B * T, C, C);

    attn_kernel<<<dim3(T / 64, B * H), 128, 0, stream>>>(qb, kb, vb, mask, ob);

    gemm_bf16_kernel<true><<<gg, 256, 0, stream>>>(ob, wob, d_out, B * T, C, C);
}
